// MultiHeadAttention_6098853560372
// MI455X (gfx1250) — matrix operand order repaired and checked by reading, not compiled, not run on hardware
//
#include <hip/hip_runtime.h>
#include <hip/hip_bf16.h>

#define D_MODEL 768
#define SEQ     2048
#define BATCH   4
#define NH      12
#define DK      64
#define ROWS    (BATCH*SEQ)   // 8192

typedef _Float16 half8  __attribute__((ext_vector_type(8)));
typedef _Float16 half16 __attribute__((ext_vector_type(16)));
typedef float    f32x8  __attribute__((ext_vector_type(8)));

// ---- WMMA fragment helpers (wave32, v_wmma_f32_16x16x32_f16) ----
// A order used here (16x32, same as B): lane L holds row M=L%16; elements K = e + 16*half
// => two contiguous 8-halfs at (k0+16*half) and (k0+16*half+8).
__device__ __forceinline__ half16 load_a_frag(const _Float16* p) {
  half8 lo = *(const half8*)(p);
  half8 hi = *(const half8*)(p + 8);
  return __builtin_shufflevector(lo, hi, 0,1,2,3,4,5,6,7,8,9,10,11,12,13,14,15);
}
// B order used here (32x16, same as A): lane L holds col N=L%16; elements K = e + 16*half
// => one contiguous 16-half (32B) load at (k0 + 16*half).
__device__ __forceinline__ half16 load_b_frag(const _Float16* p) {
  return *(const half16*)(p);
}
__device__ __forceinline__ f32x8 wmma16(half16 a, half16 b, f32x8 c) {
  return __builtin_amdgcn_wmma_f32_16x16x32_f16(false, a, false, b, (short)0, c,
                                                false, false);
}

// Async global->LDS copy of one 16B chunk (ASYNCcnt-tracked DMA path).
__device__ __forceinline__ void async_copy_b128(const void* gp, void* lp) {
  unsigned lds_off = (unsigned)(uintptr_t)lp;   // low 32 bits of generic = LDS addr
  asm volatile("global_load_async_to_lds_b128 %0, %1, off"
               :: "v"(lds_off), "v"(gp) : "memory");
}
__device__ __forceinline__ void async_wait0() {
  asm volatile("s_wait_asynccnt 0" ::: "memory");
}

// ---- Kernel 1: cast x (f32 -> f16) ----
__global__ void cast_f32_to_f16(const float* __restrict__ src,
                                _Float16* __restrict__ dst, int n) {
  int i = blockIdx.x * blockDim.x + threadIdx.x;
  int stride = gridDim.x * blockDim.x;
  for (; i < n; i += stride) dst[i] = (_Float16)src[i];
}

// ---- Kernel 2: cast + transpose 768x768 weight: WT[n*768+k] = (f16)W[k*768+n] ----
__global__ void cast_transpose_w(const float* __restrict__ W,
                                 _Float16* __restrict__ WT) {
  int i = blockIdx.x * blockDim.x + threadIdx.x;
  int stride = gridDim.x * blockDim.x;
  const int total = D_MODEL * D_MODEL;
  for (; i < total; i += stride) {
    int n = i / D_MODEL, k = i % D_MODEL;
    WT[i] = (_Float16)W[k * D_MODEL + n];
  }
}

// ---- Kernel 3: fused QKV projection ----
// One 256-thread workgroup owns a 16-row x 768-col output stripe of one of
// Q/K/V.  The 16x768 A panel (24 KB) is async-DMA'd into LDS once and shared
// by all 8 waves; wave w computes N-tiles w*6 .. w*6+5, reusing each LDS A
// fragment 6x.  All 6 B fragments are loaded before the 6 WMMAs of each
// k-step so the scheduler can clause the loads and wait incrementally.
__global__ __launch_bounds__(256) void qkv_proj(
    const _Float16* __restrict__ xh,
    const _Float16* __restrict__ wqT, const _Float16* __restrict__ wkT,
    const _Float16* __restrict__ wvT,
    const float* __restrict__ bq, const float* __restrict__ bk,
    const float* __restrict__ bv,
    _Float16* __restrict__ Qh, _Float16* __restrict__ Kh,
    _Float16* __restrict__ Vt) {
  __shared__ __attribute__((aligned(32))) _Float16 Alds[16 * D_MODEL];  // 24 KB

  const int tid   = threadIdx.x;
  const int which = blockIdx.x / (ROWS / 16);   // 0..2
  const int mt    = blockIdx.x % (ROWS / 16);   // 0..511

  // ---- async-stage A panel: 16 rows x 768 f16 = 1536 x 16B chunks ----
  const _Float16* Abase = xh + (size_t)mt * 16 * D_MODEL;
#pragma unroll
  for (int i = 0; i < 6; ++i) {
    int c = tid + i * 256;                      // 0..1535, uniform trip count
    async_copy_b128((const char*)Abase + c * 16, (char*)Alds + c * 16);
  }
  async_wait0();
  __syncthreads();

  const int lane = tid & 31, wid = tid >> 5;
  const int half = lane >> 4, ln = lane & 15;

  const _Float16* WT  = (which == 0) ? wqT : (which == 1) ? wkT : wvT;
  const float*    bia = (which == 0) ? bq  : (which == 1) ? bk  : bv;

  const _Float16* arow = Alds + ln * D_MODEL + 16 * half;   // LDS A fragments
  f32x8 c0 = {}, c1 = {}, c2 = {}, c3 = {}, c4 = {}, c5 = {};

#pragma unroll 2
  for (int kk = 0; kk < D_MODEL; kk += 32) {
    const _Float16* bcol =
        WT + (size_t)(wid * 6 * 16 + ln) * D_MODEL + 16 * half + kk;
    // issue all loads first (clause-friendly, incremental waits)
    half16 a  = load_a_frag(arow + kk);                    // ds_load_b128 x2
    half16 b0 = load_b_frag(bcol + 0 * 16 * D_MODEL);
    half16 b1 = load_b_frag(bcol + 1 * 16 * D_MODEL);
    half16 b2 = load_b_frag(bcol + 2 * 16 * D_MODEL);
    half16 b3 = load_b_frag(bcol + 3 * 16 * D_MODEL);
    half16 b4 = load_b_frag(bcol + 4 * 16 * D_MODEL);
    half16 b5 = load_b_frag(bcol + 5 * 16 * D_MODEL);
    c0 = wmma16(a, b0, c0);
    c1 = wmma16(a, b1, c1);
    c2 = wmma16(a, b2, c2);
    c3 = wmma16(a, b3, c3);
    c4 = wmma16(a, b4, c4);
    c5 = wmma16(a, b5, c5);
  }

  f32x8 acc[6] = {c0, c1, c2, c3, c4, c5};
#pragma unroll
  for (int j = 0; j < 6; ++j) {
    const int ng = (wid * 6 + j) * 16 + ln;     // global col (C layout: lane=N)
    const float bias = bia[ng];
    const int h = ng / DK, d = ng % DK;
#pragma unroll
    for (int r = 0; r < 8; ++r) {
      int mg = mt * 16 + r + 8 * half;          // global row
      int b_ = mg / SEQ, s = mg % SEQ;
      _Float16 val = (_Float16)(acc[j][r] + bias);
      if (which == 2)
        Vt[((size_t)(b_ * NH + h) * DK + d) * SEQ + s] = val;
      else {
        _Float16* dst = (which == 0) ? Qh : Kh;
        dst[((size_t)(b_ * NH + h) * SEQ + s) * DK + d] = val;
      }
    }
  }
}

// ---- Kernel 4: causal flash attention. One wave per (b,h,qtile). ----
__global__ __launch_bounds__(32) void attention(
    const _Float16* __restrict__ Qh, const _Float16* __restrict__ Kh,
    const _Float16* __restrict__ Vt, _Float16* __restrict__ Oh) {
  __shared__ __attribute__((aligned(32))) _Float16 P[16 * 32];  // 1 KB P-tile

  const int lane = threadIdx.x & 31;
  const int half = lane >> 4, ln = lane & 15;
  const int bh = blockIdx.y;              // 0..47 = b*NH+h
  const int qt = blockIdx.x;              // 0..127
  const int qs = qt * 16;
  const int b = bh / NH, h = bh % NH;

  const _Float16* Qbase = Qh + (size_t)bh * SEQ * DK;
  const _Float16* Kbase = Kh + (size_t)bh * SEQ * DK;
  const _Float16* Vbase = Vt + (size_t)bh * DK * SEQ;

  half16 aq0 = load_a_frag(Qbase + (size_t)(qs + ln) * DK + 16 * half);
  half16 aq1 = load_a_frag(Qbase + (size_t)(qs + ln) * DK + 32 + 16 * half);

  f32x8 o0 = {}, o1 = {}, o2 = {}, o3 = {};
  float mrow[8], lrow[8];
  const float NEG_INF = -__builtin_inff();
#pragma unroll
  for (int r = 0; r < 8; ++r) { mrow[r] = NEG_INF; lrow[r] = 0.0f; }

  const float L2E   = 1.4426950408889634f;
  const float SCALE = 0.125f;  // 1/sqrt(64)

  for (int kb = 0; kb <= qs + 15; kb += 32) {
    // ---- scores for keys kb..kb+31: 4 WMMAs ----
    half16 bk0a = load_b_frag(Kbase + (size_t)(kb + ln) * DK + 16 * half);
    half16 bk0b = load_b_frag(Kbase + (size_t)(kb + ln) * DK + 32 + 16 * half);
    half16 bk1a = load_b_frag(Kbase + (size_t)(kb + 16 + ln) * DK + 16 * half);
    half16 bk1b = load_b_frag(Kbase + (size_t)(kb + 16 + ln) * DK + 32 + 16 * half);
    f32x8 s0 = {}, s1 = {};
    s0 = wmma16(aq0, bk0a, s0); s0 = wmma16(aq1, bk0b, s0);
    s1 = wmma16(aq0, bk1a, s1); s1 = wmma16(aq1, bk1b, s1);

    // ---- online softmax (C layout: lane=key n, VGPR r => query m=r+8*half) ----
#pragma unroll
    for (int r = 0; r < 8; ++r) {
      int q = qs + r + 8 * half;
      float v0 = s0[r] * SCALE; if (kb + ln > q)      v0 = NEG_INF;
      float v1 = s1[r] * SCALE; if (kb + 16 + ln > q) v1 = NEG_INF;
      float mx = fmaxf(v0, v1);
      for (int off = 1; off < 16; off <<= 1)
        mx = fmaxf(mx, __shfl_xor(mx, off, 32));      // reduce within 16-lane half
      float mnew  = fmaxf(mrow[r], mx);
      float alpha = exp2f((mrow[r] - mnew) * L2E);
      float p0 = exp2f((v0 - mnew) * L2E);
      float p1 = exp2f((v1 - mnew) * L2E);
      float rs = p0 + p1;
      for (int off = 1; off < 16; off <<= 1)
        rs += __shfl_xor(rs, off, 32);
      lrow[r] = lrow[r] * alpha + rs;
      mrow[r] = mnew;
      o0[r] *= alpha; o1[r] *= alpha; o2[r] *= alpha; o3[r] *= alpha;
      int m = r + 8 * half;
      P[m * 32 + ln]      = (_Float16)p0;             // C-layout -> row-major P
      P[m * 32 + 16 + ln] = (_Float16)p1;
    }
    __syncthreads();   // single-wave WG: orders LDS store->load

    // ---- reload P as A fragment (16x32) and do P @ V: 4 WMMAs ----
    half16 ap = load_a_frag(&P[ln * 32 + 16 * half]);
    half16 bv0 = load_b_frag(Vbase + (size_t)(0 * 16 + ln) * SEQ + kb + 16 * half);
    half16 bv1 = load_b_frag(Vbase + (size_t)(1 * 16 + ln) * SEQ + kb + 16 * half);
    half16 bv2 = load_b_frag(Vbase + (size_t)(2 * 16 + ln) * SEQ + kb + 16 * half);
    half16 bv3 = load_b_frag(Vbase + (size_t)(3 * 16 + ln) * SEQ + kb + 16 * half);
    o0 = wmma16(ap, bv0, o0);
    o1 = wmma16(ap, bv1, o1);
    o2 = wmma16(ap, bv2, o2);
    o3 = wmma16(ap, bv3, o3);
    __syncthreads();
  }

  // ---- normalize and store Oh (B*S x 768 row-major f16) ----
#pragma unroll
  for (int r = 0; r < 8; ++r) {
    float inv = 1.0f / lrow[r];
    int m = qs + r + 8 * half;
    size_t row = ((size_t)b * SEQ + m) * D_MODEL + h * DK + ln;
    Oh[row + 0]  = (_Float16)(o0[r] * inv);
    Oh[row + 16] = (_Float16)(o1[r] * inv);
    Oh[row + 32] = (_Float16)(o2[r] * inv);
    Oh[row + 48] = (_Float16)(o3[r] * inv);
  }
}

// ---- Kernel 5: output projection (Oh @ Wo + bo) -> fp32 ----
// Same LDS-staged stripe structure as qkv_proj.
__global__ __launch_bounds__(256) void out_proj(
    const _Float16* __restrict__ Oh, const _Float16* __restrict__ woT,
    const float* __restrict__ bo, float* __restrict__ out) {
  __shared__ __attribute__((aligned(32))) _Float16 Alds[16 * D_MODEL];  // 24 KB

  const int tid = threadIdx.x;
  const int mt  = blockIdx.x;                   // 0..511

  const _Float16* Abase = Oh + (size_t)mt * 16 * D_MODEL;
#pragma unroll
  for (int i = 0; i < 6; ++i) {
    int c = tid + i * 256;
    async_copy_b128((const char*)Abase + c * 16, (char*)Alds + c * 16);
  }
  async_wait0();
  __syncthreads();

  const int lane = tid & 31, wid = tid >> 5;
  const int half = lane >> 4, ln = lane & 15;

  const _Float16* arow = Alds + ln * D_MODEL + 16 * half;
  f32x8 c0 = {}, c1 = {}, c2 = {}, c3 = {}, c4 = {}, c5 = {};

#pragma unroll 2
  for (int kk = 0; kk < D_MODEL; kk += 32) {
    const _Float16* bcol =
        woT + (size_t)(wid * 6 * 16 + ln) * D_MODEL + 16 * half + kk;
    half16 a  = load_a_frag(arow + kk);
    half16 b0 = load_b_frag(bcol + 0 * 16 * D_MODEL);
    half16 b1 = load_b_frag(bcol + 1 * 16 * D_MODEL);
    half16 b2 = load_b_frag(bcol + 2 * 16 * D_MODEL);
    half16 b3 = load_b_frag(bcol + 3 * 16 * D_MODEL);
    half16 b4 = load_b_frag(bcol + 4 * 16 * D_MODEL);
    half16 b5 = load_b_frag(bcol + 5 * 16 * D_MODEL);
    c0 = wmma16(a, b0, c0);
    c1 = wmma16(a, b1, c1);
    c2 = wmma16(a, b2, c2);
    c3 = wmma16(a, b3, c3);
    c4 = wmma16(a, b4, c4);
    c5 = wmma16(a, b5, c5);
  }

  f32x8 acc[6] = {c0, c1, c2, c3, c4, c5};
#pragma unroll
  for (int j = 0; j < 6; ++j) {
    const int ng = (wid * 6 + j) * 16 + ln;
    const float bias = bo[ng];
#pragma unroll
    for (int r = 0; r < 8; ++r) {
      int mg = mt * 16 + r + 8 * half;
      out[(size_t)mg * D_MODEL + ng] = acc[j][r] + bias;
    }
  }
}

extern "C" void kernel_launch(void* const* d_in, const int* in_sizes, int n_in,
                              void* d_out, int out_size, void* d_ws, size_t ws_size,
                              hipStream_t stream) {
  const float* x  = (const float*)d_in[0];
  const float* Wq = (const float*)d_in[1];
  const float* bq = (const float*)d_in[2];
  const float* Wk = (const float*)d_in[3];
  const float* bk = (const float*)d_in[4];
  const float* Wv = (const float*)d_in[5];
  const float* bv = (const float*)d_in[6];
  const float* Wo = (const float*)d_in[7];
  const float* bo = (const float*)d_in[8];
  float* out = (float*)d_out;

  // f16 workspace carve-out (~52 MB): Oh reuses the xh buffer.
  _Float16* ws = (_Float16*)d_ws;
  size_t off = 0;
  _Float16* xh  = ws + off; off += (size_t)ROWS * D_MODEL;
  _Float16* wqT = ws + off; off += (size_t)D_MODEL * D_MODEL;
  _Float16* wkT = ws + off; off += (size_t)D_MODEL * D_MODEL;
  _Float16* wvT = ws + off; off += (size_t)D_MODEL * D_MODEL;
  _Float16* woT = ws + off; off += (size_t)D_MODEL * D_MODEL;
  _Float16* Qh  = ws + off; off += (size_t)ROWS * D_MODEL;
  _Float16* Kh  = ws + off; off += (size_t)ROWS * D_MODEL;
  _Float16* Vt  = ws + off; off += (size_t)ROWS * D_MODEL;
  _Float16* Oh  = xh;  // xh dead after qkv_proj

  cast_f32_to_f16<<<4096, 256, 0, stream>>>(x, xh, ROWS * D_MODEL);
  cast_transpose_w<<<1152, 256, 0, stream>>>(Wq, wqT);
  cast_transpose_w<<<1152, 256, 0, stream>>>(Wk, wkT);
  cast_transpose_w<<<1152, 256, 0, stream>>>(Wv, wvT);
  cast_transpose_w<<<1152, 256, 0, stream>>>(Wo, woT);

  // 3 outputs * 512 row-stripes, one workgroup per stripe
  qkv_proj<<<1536, 256, 0, stream>>>(xh, wqT, wkT, wvT, bq, bk, bv, Qh, Kh, Vt);

  dim3 agrid(SEQ / 16, BATCH * NH);   // (128, 48)
  attention<<<agrid, 32, 0, stream>>>(Qh, Kh, Vt, Oh);

  out_proj<<<512, 256, 0, stream>>>(Oh, woT, bo, out);
}
